// Decoder_901943132844
// MI455X (gfx1250) — matrix operand order repaired and checked by reading, not compiled, not run on hardware
//
#include <hip/hip_runtime.h>

// ---------------------------------------------------------------------------
// MI455X (gfx1250) attention-LSTM decoder.
//
// Perf reasoning (no runtime available; from specs):
//  * Attention re-reads keys+values every step: 262 MB/step in f32 -> 65 GB
//    over 250 steps (~2.8 ms at 23.3 TB/s HBM). One-time bf16 conversion
//    halves traffic AND makes keys+values (131 MB) resident in the 192 MB L2,
//    so the steady-state scan streams from L2.
//  * All GEMMs (LSTM gates, projection) run on v_wmma_f32_16x16x32_bf16
//    (f32 accumulate), tiles 16x16, K unrolled by 32, fragment layouts per
//    CDNA5 ISA 7.12.2. Gate matrices are staged in dynamic LDS (16 x 4H f32,
//    max 128 KB < 320 KB/WGP) so the LSTM cell nonlinearity fuses into the
//    same kernel (each block owns 16 batch rows end-to-end -> no extra sync).
//  * Attention energy/context are per-row mat-vecs (keys index both output
//    dims -> not GEMM-shaped); done with 128-bit bf16 loads + f32 FMA and
//    global_prefetch_b8 on the streaming operands.
//
// Workspace use (~141 MB): bf16 keys/values/weights + f32 cell states.
// ---------------------------------------------------------------------------

#define DEV __device__ __forceinline__

typedef __attribute__((ext_vector_type(16))) __bf16 v16bf;
typedef __attribute__((ext_vector_type(8)))  __bf16 v8bf;
typedef __attribute__((ext_vector_type(8)))  float  v8f;
typedef __attribute__((ext_vector_type(4)))  unsigned int u32x4;

union ABFrag { u32x4 u[2]; v16bf v; };
union BFrag8 { u32x4 u;    v8bf  v; };

constexpr int T_ = 1000, N_ = 256, KS_ = 128, VS_ = 128;
constexpr int HID_ = 512, EMB_ = 256, MAXLEN_ = 250, VOCAB_ = 30;

DEV float sigf(float x) { return 1.0f / (1.0f + __expf(-x)); }

// One 16x16 output tile, K-loop by 32 with v_wmma_f32_16x16x32_bf16.
// xrow = X + row*ldx + hi*8   (A fragment base for this lane)
// wrow = W + col*ldw + hi*8   (B fragment base for this lane; B = W^T tile)
DEV v8f wmma_seg(v8f acc, const __bf16* xrow, const __bf16* wrow, int K) {
  for (int kb = 0; kb < K; kb += 32) {
    ABFrag a, b;
    a.u[0] = *(const u32x4*)(xrow + kb);        // K = kb+{0..7}  (or +8..15 for hi)
    a.u[1] = *(const u32x4*)(xrow + kb + 16);   // K = kb+{16..23} (or +24..31)
    b.u[0] = *(const u32x4*)(wrow + kb);        // K = kb+{0..7}  (or +8..15 for hi)
    b.u[1] = *(const u32x4*)(wrow + kb + 16);   // K = kb+{16..23} (or +24..31)
    acc = __builtin_amdgcn_wmma_f32_16x16x32_bf16(
        false, a.v, false, b.v, (short)0, acc, false, false);
  }
  return acc;
}

// ---------------------------------------------------------------------------
// Fused GEMM (gates = [XA|XB] @ Wih^T + XC @ Whh^T) + LSTM cell.
// Block = 256 threads (8 waves) owns 16 batch rows and ALL 4H gate columns;
// gate tile results staged in dynamic LDS, then cell applied after barrier.
// If `text` != nullptr, segment A rows are gathered from the bf16 embedding
// table via the token index (fuses the embedding lookup into LSTM1).
// ---------------------------------------------------------------------------
template <int H>
__global__ __launch_bounds__(256) void lstm_layer_kernel(
    const __bf16* XA, int ldxa, int Ka,
    const __bf16* XB, int ldxb, int Kb,
    const __bf16* XC, int ldxc, int Kc,
    const __bf16* __restrict__ Wih, int ldwih,  // (4H, Ka+Kb) row-major bf16
    const __bf16* __restrict__ Whh, int ldwhh,  // (4H, Kc)    row-major bf16
    const float* __restrict__ b_ih, const float* __restrict__ b_hh,
    float* c_state, __bf16* h_out,
    const int* __restrict__ text, int step, const __bf16* __restrict__ embW) {
  extern __shared__ float gs[];  // [16][4H]
  const int H4 = 4 * H;
  const int tid = threadIdx.x;
  const int lane = tid & 31, wave = tid >> 5;
  const int l = lane & 15, hi = lane >> 4;
  const int n0 = blockIdx.x * 16;

  // Per-lane A-operand row bases (lane l covers batch row n0+l).
  const __bf16* xrowA;
  if (text) {
    int tok = text[(n0 + l) * MAXLEN_ + step];
    xrowA = embW + (size_t)tok * EMB_ + hi * 8;
  } else {
    xrowA = XA + (size_t)(n0 + l) * ldxa + hi * 8;
  }
  const __bf16* xrowB = (Kb > 0) ? (XB + (size_t)(n0 + l) * ldxb + hi * 8) : nullptr;
  const __bf16* xrowC = XC + (size_t)(n0 + l) * ldxc + hi * 8;

  const int tilesPerWave = H4 / 16 / 8;  // 16 (H=512) or 4 (H=128)
  for (int tt = 0; tt < tilesPerWave; ++tt) {
    const int o = (wave * tilesPerWave + tt) * 16;
    v8f acc = {0.f, 0.f, 0.f, 0.f, 0.f, 0.f, 0.f, 0.f};
    const __bf16* wrowI = Wih + (size_t)(o + l) * ldwih + hi * 8;
    acc = wmma_seg(acc, xrowA, wrowI, Ka);
    if (Kb > 0) acc = wmma_seg(acc, xrowB, wrowI + Ka, Kb);
    const __bf16* wrowH = Whh + (size_t)(o + l) * ldwhh + hi * 8;
    acc = wmma_seg(acc, xrowC, wrowH, Kc);
    // C/D layout: vgpr i -> row (i + hi*8), col = l (within tile).
#pragma unroll
    for (int i = 0; i < 8; ++i) gs[(i + hi * 8) * H4 + o + l] = acc[i];
  }
  __syncthreads();

  // LSTM cell: gates order i,f,g,o along columns.
  for (int e = tid; e < 16 * H; e += 256) {
    const int r = e / H, j = e - r * H;
    const int n = n0 + r;
    float gi = gs[r * H4 + j]           + b_ih[j]           + b_hh[j];
    float gf = gs[r * H4 + H + j]       + b_ih[H + j]       + b_hh[H + j];
    float gg = gs[r * H4 + 2 * H + j]   + b_ih[2 * H + j]   + b_hh[2 * H + j];
    float go = gs[r * H4 + 3 * H + j]   + b_ih[3 * H + j]   + b_hh[3 * H + j];
    float c  = c_state[(size_t)n * H + j];
    float cn = sigf(gf) * c + sigf(gi) * tanhf(gg);
    float hn = sigf(go) * tanhf(cn);
    c_state[(size_t)n * H + j] = cn;
    h_out[(size_t)n * H + j]   = (__bf16)hn;
  }
}

// ---------------------------------------------------------------------------
// Attention: one block (256 threads) per batch row n.
// energy -> masked softmax -> attn row written to d_out -> context (bf16).
// keys/values streamed as bf16 (L2-resident after step 0).
// ---------------------------------------------------------------------------
__global__ __launch_bounds__(256) void attention_kernel(
    const __bf16* __restrict__ keysb, const __bf16* __restrict__ valsb,
    const int* __restrict__ seqlen, const __bf16* __restrict__ h3b,
    float* __restrict__ attn_out, __bf16* __restrict__ ctxb, int step) {
  __shared__ float h3s[KS_];
  __shared__ float red[256];
  __shared__ float cpart[8][8];
  const int n = blockIdx.x, tid = threadIdx.x;
  const int lane = tid & 31, wave = tid >> 5;

  if (tid < KS_) h3s[tid] = (float)h3b[(size_t)n * KS_ + tid];
  __syncthreads();

  const int L = seqlen[n];
  float* arow = attn_out + ((size_t)step * N_ + n) * T_;

  // Phase 1: energies + max.
  float lmax = -3.0e38f;
  for (int t = tid; t < T_; t += 256) {
    const __bf16* kp = keysb + ((size_t)t * N_ + n) * KS_;
    if (t + 256 < T_)
      __builtin_prefetch(keysb + ((size_t)(t + 256) * N_ + n) * KS_, 0, 0);
    float e = 0.f;
#pragma unroll
    for (int k0 = 0; k0 < KS_; k0 += 8) {
      BFrag8 b; b.u = *(const u32x4*)(kp + k0);
#pragma unroll
      for (int j = 0; j < 8; ++j) e = fmaf((float)b.v[j], h3s[k0 + j], e);
    }
    if (t >= L) e = -1.0e9f;
    arow[t] = e;
    lmax = fmaxf(lmax, e);
  }
  red[tid] = lmax; __syncthreads();
  for (int s = 128; s > 0; s >>= 1) { if (tid < s) red[tid] = fmaxf(red[tid], red[tid + s]); __syncthreads(); }
  const float m = red[0];
  __syncthreads();

  // Phase 2: exp + sum (exp values kept in registers: <=4 per thread).
  float pv[4];
  float lsum = 0.f;
  {
    int idx = 0;
    for (int t = tid; t < T_; t += 256, ++idx) {
      float p = __expf(arow[t] - m);
      pv[idx] = p;
      lsum += p;
    }
  }
  red[tid] = lsum; __syncthreads();
  for (int s = 128; s > 0; s >>= 1) { if (tid < s) red[tid] += red[tid + s]; __syncthreads(); }
  const float inv = 1.0f / red[0];
  __syncthreads();
  {
    int idx = 0;
    for (int t = tid; t < T_; t += 256, ++idx) {
      pv[idx] *= inv;
      arow[t] = pv[idx];  // final attention weights -> output
    }
  }

  // Phase 3: context = sum_t attn[t] * values[t,n,:], 8 columns at a time.
  for (int v0 = 0; v0 < VS_; v0 += 8) {
    float acc[8];
#pragma unroll
    for (int j = 0; j < 8; ++j) acc[j] = 0.f;
    int idx = 0;
    for (int t = tid; t < T_; t += 256, ++idx) {
      const float w = pv[idx];
      const __bf16* vp = valsb + ((size_t)t * N_ + n) * VS_ + v0;
      if (t + 256 < T_)
        __builtin_prefetch(valsb + ((size_t)(t + 256) * N_ + n) * VS_ + v0, 0, 0);
      BFrag8 b; b.u = *(const u32x4*)vp;
#pragma unroll
      for (int j = 0; j < 8; ++j) acc[j] = fmaf(w, (float)b.v[j], acc[j]);
    }
#pragma unroll
    for (int j = 0; j < 8; ++j)
      for (int off = 16; off > 0; off >>= 1) acc[j] += __shfl_down(acc[j], off, 32);
    if (lane == 0) {
#pragma unroll
      for (int j = 0; j < 8; ++j) cpart[wave][j] = acc[j];
    }
    __syncthreads();
    if (tid < 8) {
      float s = 0.f;
#pragma unroll
      for (int w = 0; w < 8; ++w) s += cpart[w][tid];
      ctxb[(size_t)n * VS_ + v0 + tid] = (__bf16)s;
    }
    __syncthreads();
  }
}

// ---------------------------------------------------------------------------
// pred = [h3|ctx] @ lin_W^T + lin_b (WMMA), then logits = pred @ char_W^T
// (VOCAB=30, tiny: scalar f32). Block = 256 threads, 16 batch rows.
// ---------------------------------------------------------------------------
__global__ __launch_bounds__(256) void pred_kernel(
    const __bf16* __restrict__ h3b, const __bf16* __restrict__ ctxb,
    const __bf16* __restrict__ linwb, const float* __restrict__ lin_b,
    const float* __restrict__ charW, const float* __restrict__ char_b,
    float* __restrict__ preds_out, int step) {
  __shared__ float ps[16 * KS_];
  const int tid = threadIdx.x;
  const int lane = tid & 31, wave = tid >> 5;
  const int l = lane & 15, hi = lane >> 4;
  const int n0 = blockIdx.x * 16;

  {  // 8 waves x 1 tile = all 128 output columns
    const int o = wave * 16;
    v8f acc = {0.f, 0.f, 0.f, 0.f, 0.f, 0.f, 0.f, 0.f};
    const __bf16* xrowA = h3b  + (size_t)(n0 + l) * KS_ + hi * 8;
    const __bf16* xrowB = ctxb + (size_t)(n0 + l) * VS_ + hi * 8;
    const __bf16* wrow  = linwb + (size_t)(o + l) * (KS_ + VS_) + hi * 8;
    acc = wmma_seg(acc, xrowA, wrow, KS_);
    acc = wmma_seg(acc, xrowB, wrow + KS_, VS_);
#pragma unroll
    for (int i = 0; i < 8; ++i)
      ps[(i + hi * 8) * KS_ + o + l] = acc[i] + lin_b[o + l];
  }
  __syncthreads();

  for (int e = tid; e < 16 * VOCAB_; e += 256) {
    const int r = e / VOCAB_, c = e - r * VOCAB_;
    float s = char_b[c];
    const float* wr = charW + (size_t)c * KS_;
#pragma unroll 8
    for (int k = 0; k < KS_; ++k) s = fmaf(ps[r * KS_ + k], wr[k], s);
    preds_out[((size_t)(n0 + r) * MAXLEN_ + step) * VOCAB_ + c] = s;
  }
}

// ---------------------------------------------------------------------------
// Housekeeping kernels.
// ---------------------------------------------------------------------------
__global__ void cvt_bf16_kernel(const float* __restrict__ src,
                                __bf16* __restrict__ dst, size_t n) {
  size_t i = (size_t)blockIdx.x * blockDim.x + threadIdx.x;
  if (i < n) dst[i] = (__bf16)src[i];
}

__global__ void init_state_kernel(const float* __restrict__ values_last,
                                  float* c1, float* c2, float* c3,
                                  __bf16* h1, __bf16* h2, __bf16* h3,
                                  __bf16* ctxb) {
  int i = blockIdx.x * blockDim.x + threadIdx.x;
  if (i < N_ * HID_) {
    c1[i] = 0.f; c2[i] = 0.f;
    h1[i] = (__bf16)0.f; h2[i] = (__bf16)0.f;
  }
  if (i < N_ * KS_) {
    c3[i] = 0.f;
    h3[i] = (__bf16)0.f;
    ctxb[i] = (__bf16)values_last[i];  // ctx_0 = values[T-1]
  }
}

// ---------------------------------------------------------------------------
extern "C" void kernel_launch(void* const* d_in, const int* in_sizes, int n_in,
                              void* d_out, int out_size, void* d_ws, size_t ws_size,
                              hipStream_t stream) {
  const float* keys   = (const float*)d_in[0];
  const float* values = (const float*)d_in[1];
  const int*   seqlen = (const int*)d_in[2];
  const int*   text   = (const int*)d_in[3];
  const float* embW   = (const float*)d_in[4];
  const float* w_ih1  = (const float*)d_in[5];
  const float* w_hh1  = (const float*)d_in[6];
  const float* b_ih1  = (const float*)d_in[7];
  const float* b_hh1  = (const float*)d_in[8];
  const float* w_ih2  = (const float*)d_in[9];
  const float* w_hh2  = (const float*)d_in[10];
  const float* b_ih2  = (const float*)d_in[11];
  const float* b_hh2  = (const float*)d_in[12];
  const float* w_ih3  = (const float*)d_in[13];
  const float* w_hh3  = (const float*)d_in[14];
  const float* b_ih3  = (const float*)d_in[15];
  const float* b_hh3  = (const float*)d_in[16];
  const float* lin_W  = (const float*)d_in[17];
  const float* lin_b  = (const float*)d_in[18];
  const float* char_W = (const float*)d_in[19];
  const float* char_b = (const float*)d_in[20];

  float* preds = (float*)d_out;
  float* attns = preds + (size_t)N_ * MAXLEN_ * VOCAB_;

  // Workspace carving (256B-aligned), ~141 MB total.
  char* w = (char*)d_ws;
  auto carve = [&](size_t bytes) -> char* {
    char* p = w;
    w += (bytes + 255) & ~(size_t)255;
    return p;
  };
  __bf16* keysb  = (__bf16*)carve((size_t)T_ * N_ * KS_ * 2);
  __bf16* valsb  = (__bf16*)carve((size_t)T_ * N_ * VS_ * 2);
  __bf16* wih1b  = (__bf16*)carve((size_t)4 * HID_ * (EMB_ + KS_) * 2);
  __bf16* whh1b  = (__bf16*)carve((size_t)4 * HID_ * HID_ * 2);
  __bf16* wih2b  = (__bf16*)carve((size_t)4 * HID_ * HID_ * 2);
  __bf16* whh2b  = (__bf16*)carve((size_t)4 * HID_ * HID_ * 2);
  __bf16* wih3b  = (__bf16*)carve((size_t)4 * KS_ * HID_ * 2);
  __bf16* whh3b  = (__bf16*)carve((size_t)4 * KS_ * KS_ * 2);
  __bf16* linwb  = (__bf16*)carve((size_t)KS_ * (KS_ + VS_) * 2);
  __bf16* embwb  = (__bf16*)carve((size_t)VOCAB_ * EMB_ * 2);
  __bf16* h1b    = (__bf16*)carve((size_t)N_ * HID_ * 2);
  __bf16* h2b    = (__bf16*)carve((size_t)N_ * HID_ * 2);
  __bf16* h3b    = (__bf16*)carve((size_t)N_ * KS_ * 2);
  __bf16* ctxb   = (__bf16*)carve((size_t)N_ * KS_ * 2);
  float*  c1     = (float*)carve((size_t)N_ * HID_ * 4);
  float*  c2     = (float*)carve((size_t)N_ * HID_ * 4);
  float*  c3     = (float*)carve((size_t)N_ * KS_ * 4);

  auto cvt = [&](const float* s, __bf16* d, size_t n) {
    int blocks = (int)((n + 255) / 256);
    cvt_bf16_kernel<<<blocks, 256, 0, stream>>>(s, d, n);
  };
  cvt(keys,   keysb, (size_t)T_ * N_ * KS_);
  cvt(values, valsb, (size_t)T_ * N_ * VS_);
  cvt(w_ih1,  wih1b, (size_t)4 * HID_ * (EMB_ + KS_));
  cvt(w_hh1,  whh1b, (size_t)4 * HID_ * HID_);
  cvt(w_ih2,  wih2b, (size_t)4 * HID_ * HID_);
  cvt(w_hh2,  whh2b, (size_t)4 * HID_ * HID_);
  cvt(w_ih3,  wih3b, (size_t)4 * KS_ * HID_);
  cvt(w_hh3,  whh3b, (size_t)4 * KS_ * KS_);
  cvt(lin_W,  linwb, (size_t)KS_ * (KS_ + VS_));
  cvt(embW,   embwb, (size_t)VOCAB_ * EMB_);

  init_state_kernel<<<(N_ * HID_ + 255) / 256, 256, 0, stream>>>(
      values + (size_t)(T_ - 1) * N_ * VS_, c1, c2, c3, h1b, h2b, h3b, ctxb);

  const size_t lds12 = 16 * (size_t)(4 * HID_) * 4;  // 128 KB
  const size_t lds3  = 16 * (size_t)(4 * KS_) * 4;   //  32 KB

  for (int step = 0; step < MAXLEN_; ++step) {
    // LSTM1: [emb(text_t) | ctx] @ w_ih1^T + h1 @ w_hh1^T
    lstm_layer_kernel<HID_><<<N_ / 16, 256, lds12, stream>>>(
        nullptr, EMB_, EMB_,
        ctxb, KS_, KS_,
        h1b, HID_, HID_,
        wih1b, EMB_ + KS_, whh1b, HID_,
        b_ih1, b_hh1, c1, h1b, text, step, embwb);
    // LSTM2: h1 @ w_ih2^T + h2 @ w_hh2^T
    lstm_layer_kernel<HID_><<<N_ / 16, 256, lds12, stream>>>(
        h1b, HID_, HID_,
        nullptr, 0, 0,
        h2b, HID_, HID_,
        wih2b, HID_, whh2b, HID_,
        b_ih2, b_hh2, c2, h2b, nullptr, step, nullptr);
    // LSTM3: h2 @ w_ih3^T + h3 @ w_hh3^T
    lstm_layer_kernel<KS_><<<N_ / 16, 256, lds3, stream>>>(
        h2b, HID_, HID_,
        nullptr, 0, 0,
        h3b, KS_, KS_,
        wih3b, HID_, whh3b, KS_,
        b_ih3, b_hh3, c3, h3b, nullptr, step, nullptr);
    // Attention (writes attn row to d_out, updates ctx)
    attention_kernel<<<N_, 256, 0, stream>>>(
        keysb, valsb, seqlen, h3b, attns, ctxb, step);
    // Projection + logits (writes preds to d_out)
    pred_kernel<<<N_ / 16, 256, 0, stream>>>(
        h3b, ctxb, linwb, lin_b, char_W, char_b, preds, step);
  }
}
